// GCBlock9_23545010717066
// MI455X (gfx1250) — matrix operand order repaired and checked by reading, not compiled, not run on hardware
//
#include <hip/hip_runtime.h>
#include <math.h>

typedef __attribute__((ext_vector_type(16))) _Float16 v16h;
typedef __attribute__((ext_vector_type(8)))  float    v8f;

#define QN 100
#define B_ 2
#define C_ 256
#define BC 512          // B*C rows
#define H_ 96
#define W_ 96
#define HW 9216         // H*W

// workspace layout (float offsets; every section 32B-aligned)
#define WS_GAP   0
#define WS_X     (WS_GAP + BC)            // x   (2,1024)
#define WS_DYN   (WS_X + B_*1024)         // dyn (2,768)
#define WS_COEF  (WS_DYN + B_*768)        // coef(512,27) = f9|w9|e9
#define WS_DIFF  (WS_COEF + BC*27)        // diff(2,256,96,96)
#define WS_PA1   (WS_DIFF + (size_t)BC*HW)        // A frags for lin:  4*32*32 v16h
#define WS_PB1   (WS_PA1 + 4*32*32*8)             // B frags for lin:  4*2*32  v16h
#define WS_PA2   (WS_PB1 + 4*2*32*8)              // A frags for q:    32*32   v16h
#define WS_PB2   (WS_PA2 + 32*32*8)               // B frags for q:    7*32    v16h

__device__ __forceinline__ float sigmoidf_(float x){ return 1.0f/(1.0f+__expf(-x)); }

// fragment element -> K mapping for 16-bit A 16x32 (ISA 7.12.2)
__device__ __forceinline__ int a_klocal(int e, int half){
    return (e < 8) ? (half*8 + e) : (16 + half*8 + (e - 8));
}

// ---------------- gap = mean(value, H, W) : one block per (b,c) ----------------
__global__ void k_gap(const float* __restrict__ value, float* __restrict__ gap){
    __shared__ float s[256];
    const int bc = blockIdx.x;
    const float* p = value + (size_t)bc*HW;
    float acc = 0.f;
    for (int i = threadIdx.x; i < HW; i += 256) acc += p[i];
    s[threadIdx.x] = acc; __syncthreads();
    for (int o = 128; o > 0; o >>= 1){
        if (threadIdx.x < o) s[threadIdx.x] += s[threadIdx.x + o];
        __syncthreads();
    }
    if (threadIdx.x == 0) gap[bc] = s[0] * (1.0f/(float)HW);
}

// ---------------- x = relu(BN(gap @ w_dp.T + b_dp)) ----------------
__global__ void k_dp(const float* __restrict__ gap, const float* __restrict__ w_dp,
                     const float* __restrict__ b_dp, const float* __restrict__ bn_g,
                     const float* __restrict__ bn_b, const float* __restrict__ bn_rm,
                     const float* __restrict__ bn_rv, float* __restrict__ x){
    int idx = blockIdx.x*256 + threadIdx.x;
    if (idx >= B_*1024) return;
    int b = idx / 1024, j = idx & 1023;
    const float* g  = gap + b*C_;
    const float* wr = w_dp + (size_t)j*C_;
    float acc = b_dp[j];
    #pragma unroll 4
    for (int k = 0; k < C_; ++k) acc += g[k]*wr[k];
    acc = (acc - bn_rm[j]) * rsqrtf(bn_rv[j] + 1e-5f) * bn_g[j] + bn_b[j];
    x[idx] = fmaxf(acc, 0.0f);
}

// ---------------- dyn = sigmoid(x @ w_dc.T + b_dc) ----------------
__global__ void k_dc(const float* __restrict__ x, const float* __restrict__ w_dc,
                     const float* __restrict__ b_dc, float* __restrict__ dyn){
    int idx = blockIdx.x*256 + threadIdx.x;
    if (idx >= B_*768) return;
    int b = idx / 768, j = idx % 768;
    const float* xr = x + b*1024;
    const float* wr = w_dc + (size_t)j*1024;
    float acc = b_dc[j];
    #pragma unroll 4
    for (int k = 0; k < 1024; ++k) acc += xr[k]*wr[k];
    dyn[idx] = sigmoidf_(acc);
}

// ---------------- pack A fragments for lin GEMM: query^T -> f16, zero-padded ----------------
// Apack[((kc*32 + mtile)*32 + lane)*16 + e] ; one thread per element (65536)
__global__ void k_pack_a1(const float* __restrict__ query, _Float16* __restrict__ Apack){
    int idx = blockIdx.x*256 + threadIdx.x;
    int e = idx & 15, lane = (idx >> 4) & 31, mtile = (idx >> 9) & 31, kc = idx >> 14;
    int half = lane >> 4, l15 = lane & 15;
    int ka = kc*32 + a_klocal(e, half);
    int m  = mtile*16 + l15;
    float v = (ka < QN) ? query[(size_t)ka*BC + m] : 0.0f;
    Apack[idx] = (_Float16)v;
}

// ---------------- pack B fragments for lin GEMM: w_lin^T -> f16, zero-padded ----------------
// Bpack[((kc*2 + ntile)*32 + lane)*16 + e] ; 4096 elements
__global__ void k_pack_b1(const float* __restrict__ w_lin, _Float16* __restrict__ Bpack){
    int idx = blockIdx.x*256 + threadIdx.x;
    int e = idx & 15, lane = (idx >> 4) & 31, ntile = (idx >> 9) & 1, kc = idx >> 10;
    int half = lane >> 4, l15 = lane & 15;
    int kb = kc*32 + a_klocal(e, half); // 16-bit B 32x16: same K order as A fragments
    int n  = ntile*16 + l15;
    float v = (n < 27 && kb < QN) ? w_lin[n*QN + kb] : 0.0f;
    Bpack[idx] = (_Float16)v;
}

// ---------------- pack B fragments for q GEMM: w_delin^T -> f16, zero-padded ----------------
// Bpack2[(ntile*32 + lane)*16 + e] ; 3584 elements
__global__ void k_pack_b2(const float* __restrict__ w_delin, _Float16* __restrict__ Bpack2){
    int idx = blockIdx.x*256 + threadIdx.x;
    if (idx >= 7*32*16) return;
    int e = idx & 15, lane = (idx >> 4) & 31, ntile = idx >> 9;
    int half = lane >> 4, l15 = lane & 15;
    int kb = a_klocal(e, half);
    int n  = ntile*16 + l15;
    float v = (n < QN && kb < 27) ? w_delin[n*27 + kb] : 0.0f;
    Bpack2[idx] = (_Float16)v;
}

// ---------------- pack A fragments for q GEMM: coef -> f16, zero-padded ----------------
// Apack2[(mtile*32 + lane)*16 + e] ; 16384 elements
__global__ void k_pack_a2(const float* __restrict__ coef, _Float16* __restrict__ Apack2){
    int idx = blockIdx.x*256 + threadIdx.x;
    int e = idx & 15, lane = (idx >> 4) & 31, mtile = idx >> 9;
    int half = lane >> 4, l15 = lane & 15;
    int kl = a_klocal(e, half);
    int m  = mtile*16 + l15;
    float v = (kl < 27) ? coef[m*27 + kl] : 0.0f;
    Apack2[idx] = (_Float16)v;
}

// ---------------- coef = (query^T @ w_lin^T + b_lin) * dyn   via WMMA ----------------
// One wave per 16x16 tile. grid = (32 Mtiles, 2 Ntiles), block = 32 (wave32).
__global__ void k_lin_wmma(const _Float16* __restrict__ Apack, const _Float16* __restrict__ Bpack,
                           const float* __restrict__ b_lin, const float* __restrict__ dyn,
                           float* __restrict__ coef){
    const int lane  = threadIdx.x;
    const int half  = lane >> 4;
    const int l15   = lane & 15;
    const int mtile = blockIdx.x;        // 0..31
    const int ntile = blockIdx.y;        // 0..1
    const v16h* Ap = (const v16h*)Apack;
    const v16h* Bp = (const v16h*)Bpack;
    v8f c = {};
    #pragma unroll
    for (int kc = 0; kc < 4; ++kc){      // K = 100 padded to 128 (4 x 32)
        v16h a = Ap[(kc*32 + mtile)*32 + lane];
        v16h b = Bp[(kc*2  + ntile)*32 + lane];
        c = __builtin_amdgcn_wmma_f32_16x16x32_f16(false, a, false, b, (short)0, c, false, false);
    }
    const int n = ntile*16 + l15;
    if (n < 27){
        const int g = n / 9;             // 0:f9 1:w9 2:e9 -> d_f/d_w/d_e scale
        #pragma unroll
        for (int r = 0; r < 8; ++r){     // C/D layout: VGPR r -> M = r + 8*half
            int m  = mtile*16 + half*8 + r;
            int bb = m >> 8, ch = m & 255;
            float d = dyn[bb*768 + g*256 + ch];
            coef[m*27 + n] = (c[r] + b_lin[n]) * d;
        }
    }
}

// ---------------- q = coef(512x27) @ w_delin^T(27x100) + b_delin  via WMMA ----------------
// output q[n][m] -> d_out[n*512 + m]  ((100,2,256) order). grid = (32 Mtiles, 7 Ntiles).
__global__ void k_q_wmma(const _Float16* __restrict__ Apack2, const _Float16* __restrict__ Bpack2,
                         const float* __restrict__ b_delin, float* __restrict__ qout){
    const int lane  = threadIdx.x;
    const int half  = lane >> 4;
    const int l15   = lane & 15;
    const int mtile = blockIdx.x;        // 0..31
    const int ntile = blockIdx.y;        // 0..6 (N=100 padded to 112)
    const v16h* Ap = (const v16h*)Apack2;
    const v16h* Bp = (const v16h*)Bpack2;
    v16h a = Ap[mtile*32 + lane];
    v16h b = Bp[ntile*32 + lane];
    v8f c = {};
    c = __builtin_amdgcn_wmma_f32_16x16x32_f16(false, a, false, b, (short)0, c, false, false);
    const int n = ntile*16 + l15;
    if (n < QN){
        #pragma unroll
        for (int r = 0; r < 8; ++r){
            int m = mtile*16 + half*8 + r;
            qout[n*BC + m] = c[r] + b_delin[n];
        }
    }
}

// ---------------- pass 1: diff = exp(-(v - dwconv(v, f9))^2) ----------------
__global__ void k_diff(const float* __restrict__ value, const float* __restrict__ coef,
                       float* __restrict__ diff){
    __shared__ float sf[9];
    const int bc = blockIdx.y;
    if (threadIdx.x < 9) sf[threadIdx.x] = coef[bc*27 + threadIdx.x];
    __syncthreads();
    const int p = blockIdx.x*256 + threadIdx.x;     // 36 blocks x 256 = 9216
    const int h = p / W_, w = p % W_;
    const float* v = value + (size_t)bc*HW;
    float yt = 0.f;
    #pragma unroll
    for (int t = 0; t < 9; ++t){
        int hh = h + t/3 - 1, ww = w + t%3 - 1;
        bool in = ((unsigned)hh < H_) && ((unsigned)ww < W_);
        int  ni = in ? (hh*W_ + ww) : p;            // clamp to safe addr, mask value
        float msk = in ? 1.0f : 0.0f;
        yt += sf[t] * (v[ni] * msk);
    }
    float d0 = v[p] - yt;
    diff[(size_t)bc*HW + p] = __expf(-d0*d0);
}

// ---------------- pass 2: fused y_temp / y_diff1 / y_diff9 / res / res_diff ----------------
__global__ void k_fuse(const float* __restrict__ value, const float* __restrict__ coef,
                       const float* __restrict__ diff, float* __restrict__ y){
    __shared__ float sc[27];
    const int bc = blockIdx.y;
    if (threadIdx.x < 27) sc[threadIdx.x] = coef[bc*27 + threadIdx.x];
    __syncthreads();
    const int p = blockIdx.x*256 + threadIdx.x;
    const int h = p / W_, w = p % W_;
    const float* v = value + (size_t)bc*HW;
    const float* d = diff  + (size_t)bc*HW;
    const float vc = v[p];
    const float dc = d[p];
    const float dv = dc*vc;                     // (diff*value) center term
    float yt = 0.f, yd1 = 0.f, yd9 = 0.f, res = 0.f, esum = 0.f;
    #pragma unroll
    for (int t = 0; t < 9; ++t){
        const float ft = sc[t], wt = sc[9+t], et = sc[18+t];
        esum += et;
        int hh = h + t/3 - 1, ww = w + t%3 - 1;
        bool in = ((unsigned)hh < H_) && ((unsigned)ww < W_);
        int  ni = in ? (hh*W_ + ww) : p;
        float msk = in ? 1.0f : 0.0f;
        float vN = v[ni] * msk;                              // zero-pad semantics
        float dN = d[ni] * msk;
        yt  += ft * vN;                                      // y_temp (recomputed)
        float Dw = sigmoidf_(fabsf(dN*vN - dv));             // Dw tap
        yd1 += Dw * vN * wt;                                 // y_diff1 (vN=0 OOB -> 0)
        yd9 += fabsf(et) * dN;                               // y_diff9
        res += et * (vN * dN);                               // res (dwconv of v*diff)
    }
    const float yd2 = (res - dv*esum) / (yd9 + 1e-10f);      // res_diff = 1x1 conv = dv*esum
    y[(size_t)bc*HW + p] = 2.0f*yt - yd1 - yd2;
}

extern "C" void kernel_launch(void* const* d_in, const int* in_sizes, int n_in,
                              void* d_out, int out_size, void* d_ws, size_t ws_size,
                              hipStream_t stream) {
    (void)in_sizes; (void)n_in; (void)out_size; (void)ws_size;
    const float* query    = (const float*)d_in[0];
    const float* value    = (const float*)d_in[1];
    /* d_in[2] hard_sigmoid_masks: unused by the reference */
    const float* w_dp     = (const float*)d_in[3];
    const float* b_dp     = (const float*)d_in[4];
    const float* bn_g     = (const float*)d_in[5];
    const float* bn_b     = (const float*)d_in[6];
    const float* bn_rm    = (const float*)d_in[7];
    const float* bn_rv    = (const float*)d_in[8];
    const float* w_dc     = (const float*)d_in[9];
    const float* b_dc     = (const float*)d_in[10];
    const float* w_lin    = (const float*)d_in[11];
    const float* b_lin    = (const float*)d_in[12];
    const float* w_delin  = (const float*)d_in[13];
    const float* b_delin  = (const float*)d_in[14];

    float* ws   = (float*)d_ws;
    float* gap  = ws + WS_GAP;
    float* x    = ws + WS_X;
    float* dyn  = ws + WS_DYN;
    float* coef = ws + WS_COEF;
    float* diff = ws + WS_DIFF;
    _Float16* pa1 = (_Float16*)(ws + WS_PA1);
    _Float16* pb1 = (_Float16*)(ws + WS_PB1);
    _Float16* pa2 = (_Float16*)(ws + WS_PA2);
    _Float16* pb2 = (_Float16*)(ws + WS_PB2);

    float* qout = (float*)d_out;          // (100, 2, 256) = 51200 floats
    float* yout = qout + QN*BC;           // (2, 256, 96, 96)

    // small dense chain
    k_gap<<<BC, 256, 0, stream>>>(value, gap);
    k_dp <<<(B_*1024 + 255)/256, 256, 0, stream>>>(gap, w_dp, b_dp, bn_g, bn_b, bn_rm, bn_rv, x);
    k_dc <<<(B_*768  + 255)/256, 256, 0, stream>>>(x, w_dc, b_dc, dyn);

    // stage zero-padded f16 fragments, then branch-free WMMA GEMMs
    k_pack_a1<<<(4*32*32*16)/256, 256, 0, stream>>>(query, pa1);
    k_pack_b1<<<(4*2*32*16)/256,  256, 0, stream>>>(w_lin, pb1);
    k_pack_b2<<<(7*32*16 + 255)/256, 256, 0, stream>>>(w_delin, pb2);
    k_lin_wmma<<<dim3(BC/16, 2), 32, 0, stream>>>(pa1, pb1, b_lin, dyn, coef);
    k_pack_a2<<<(32*32*16)/256, 256, 0, stream>>>(coef, pa2);
    k_q_wmma  <<<dim3(BC/16, 7), 32, 0, stream>>>(pa2, pb2, b_delin, qout);

    // bandwidth-dominant fused stencil passes
    k_diff<<<dim3(HW/256, BC), 256, 0, stream>>>(value, coef, diff);
    k_fuse<<<dim3(HW/256, BC), 256, 0, stream>>>(value, coef, diff, yout);
}
